// InteractionModule_52750788329787
// MI455X (gfx1250) — matrix operand order repaired and checked by reading, not compiled, not run on hardware
//
#include <hip/hip_runtime.h>
#include <hip/hip_bf16.h>

// ---------------------------------------------------------------------------
// Problem constants (from the reference)
// ---------------------------------------------------------------------------
#define NNODES   8192
#define NEDGES   65536
#define NS       48
#define EDGESZ   64
#define SHD      9
#define NEF      224           // 2*NS + 2*EDGE_SIZE
#define HSTRIDE  224           // h feature buffer stride (max irrep dim)
#define MAXLEN   20.0f

typedef __attribute__((ext_vector_type(16))) _Float16     v16h;
typedef __attribute__((ext_vector_type(8)))  float        v8f;
typedef __attribute__((ext_vector_type(4)))  unsigned int v4u;
typedef __attribute__((ext_vector_type(8)))  unsigned int v8u;

// ---------------------------------------------------------------------------
// WMMA fragment loaders (16x16x32 f16 -> f32), layouts per CDNA5 ISA 7.12.2
// ---------------------------------------------------------------------------
__device__ __forceinline__ v16h load_a_frag(const _Float16* A, int lda,
                                            int row0, int k0, int kmax) {
  int lane = threadIdx.x & 31;
  int m = lane & 15, hi = lane >> 4;
  const _Float16* rowp = A + (size_t)(row0 + m) * lda;
  v16h a;
#pragma unroll
  for (int r = 0; r < 8; ++r) {
    int kk = (r < 4) ? (2 * r + 8 * hi) : (16 + 2 * (r - 4) + 8 * hi);
    int k = k0 + kk;
    a[2 * r]     = (k     < kmax) ? rowp[k]     : (_Float16)0.f;
    a[2 * r + 1] = (k + 1 < kmax) ? rowp[k + 1] : (_Float16)0.f;
  }
  return a;
}

__device__ __forceinline__ v16h load_b_frag(const _Float16* W, int ldn,
                                            int k0, int col0, int kmax, int nmax) {
  int lane = threadIdx.x & 31;
  int n = lane & 15, hi = lane >> 4;
  int col = col0 + n;
  bool cok = (col < nmax);
  v16h b;
#pragma unroll
  for (int j = 0; j < 16; ++j) {
    int k = k0 + ((j < 8) ? (j + 8 * hi) : (16 + 8 * hi + (j - 8)));
    b[j] = (cok && k < kmax) ? W[(size_t)k * ldn + col] : (_Float16)0.f;
  }
  return b;
}

__device__ __forceinline__ v8f wmma_f16(v16h a, v16h b, v8f c) {
  return __builtin_amdgcn_wmma_f32_16x16x32_f16(false, a, false, b,
                                                (short)0, c, false, false);
}

// ---------------------------------------------------------------------------
// Tensor Data Mover: DMA a (rows x 16) f16 column-tile of a row-major
// (rows x N) weight matrix into LDS. D# descriptor per CDNA5 ISA §8:
//   group0: count=1 | lds_addr | global_addr[56:0] | type=2
//   group1: data_size=1(2B), tensor_dim0/1, tile_dim0=16, tile_dim1=rows,
//           tensor_dim0_stride=N
//   groups 2/3: benign for a 2D tensor (higher dims unused)
// Tracked with TENSORcnt; waited via s_wait_tensorcnt.
// ---------------------------------------------------------------------------
__device__ __forceinline__ void tdm_load_w_tile(const _Float16* W, int N, int col0,
                                                unsigned lds_off, int rows) {
  unsigned long long ga = (unsigned long long)(W + col0);
  unsigned dim0 = (unsigned)(N - col0);   // remaining width (elements)
  unsigned dim1 = (unsigned)rows;
  unsigned long long str0 = (unsigned long long)(unsigned)N;
  v4u g0;
  g0[0] = 1u;                                        // count=1, user descriptor
  g0[1] = lds_off;                                   // LDS byte address
  g0[2] = (unsigned)(ga & 0xffffffffu);              // global_addr[31:0]
  g0[3] = (unsigned)((ga >> 32) & 0x1ffffffu) | (2u << 30);  // [56:32] | type=2
  v8u g1;
  g1[0] = (1u << 16);                                // data_size = 1 -> 2 bytes
  g1[1] = (dim0 & 0xffffu) << 16;                    // tensor_dim0[15:0]
  g1[2] = ((dim0 >> 16) & 0xffffu) | ((dim1 & 0xffffu) << 16);
  g1[3] = ((dim1 >> 16) & 0xffffu) | (16u << 16);    // tile_dim0 = 16
  g1[4] = (unsigned)rows;                            // tile_dim1 = rows, tile_dim2=0
  g1[5] = (unsigned)(str0 & 0xffffffffu);            // tensor_dim0_stride[31:0]
  g1[6] = (unsigned)((str0 >> 32) & 0xffffu);        // stride0 msbs | stride1 lo=0
  g1[7] = 0u;
  v4u g2; g2[0] = 1u; g2[1] = 1u; g2[2] = 0u; g2[3] = 0u;
  v4u g3; g3[0] = 0u; g3[1] = 0u; g3[2] = 0u; g3[3] = 0u;
  asm volatile("tensor_load_to_lds %0, %1, %2, %3"
               :: "s"(g0), "s"(g1), "s"(g2), "s"(g3)
               : "memory");
}

__device__ __forceinline__ void tdm_wait0() {
  __builtin_amdgcn_s_wait_tensorcnt(0);
}

// ---------------------------------------------------------------------------
// Generic wave-tiled GEMM: out = act(A[MxK] @ W[KxN] + bias)
// ---------------------------------------------------------------------------
template <bool RELU, bool OUT16>
__global__ void wgemm_kernel(const _Float16* __restrict__ A,
                             const _Float16* __restrict__ W,
                             const float* __restrict__ bias,
                             _Float16* __restrict__ out16,
                             float* __restrict__ out32,
                             int M, int N, int K) {
  int wid = (int)((blockIdx.x * blockDim.x + threadIdx.x) >> 5);
  int tN = (N + 15) >> 4;
  int tM = M >> 4;
  if (wid >= tM * tN) return;
  int mt = wid / tN, nt = wid % tN;
  int row0 = mt << 4, col0 = nt << 4;
  v8f acc = {};
  for (int k0 = 0; k0 < K; k0 += 32) {
    if (k0 + 32 < K)  // warm the weight stream (global_prefetch_b8)
      __builtin_prefetch(W + (size_t)(k0 + 32) * N + col0, 0, 1);
    v16h a = load_a_frag(A, K, row0, k0, K);
    v16h b = load_b_frag(W, N, k0, col0, K, N);
    acc = wmma_f16(a, b, acc);
  }
  int lane = threadIdx.x & 31;
  int n = lane & 15, hi = lane >> 4;
  int col = col0 + n;
  if (col >= N) return;
  float bb = bias ? bias[col] : 0.f;
#pragma unroll
  for (int r = 0; r < 8; ++r) {
    int row = row0 + r + 8 * hi;
    float v = acc[r] + bb;
    if (RELU) v = v > 0.f ? v : 0.f;
    if (OUT16) out16[(size_t)row * N + col] = (_Float16)v;
    else       out32[(size_t)row * N + col] = v;
  }
}

// ---------------------------------------------------------------------------
// f32 -> f16 conversion
// ---------------------------------------------------------------------------
__global__ void cvt16_kernel(const float* __restrict__ in,
                             _Float16* __restrict__ out, long long n) {
  long long stride = (long long)gridDim.x * blockDim.x;
  for (long long i = blockIdx.x * (long long)blockDim.x + threadIdx.x; i < n; i += stride)
    out[i] = (_Float16)in[i];
}

// ---------------------------------------------------------------------------
// Edge geometry: spherical harmonics + Gaussian radial embedding
// ---------------------------------------------------------------------------
__global__ void edge_geom_kernel(const float* __restrict__ coords,
                                 const int* __restrict__ edges,
                                 float* __restrict__ sh,
                                 _Float16* __restrict__ ge) {
  int e = blockIdx.x * blockDim.x + threadIdx.x;
  if (e >= NEDGES) return;
  int s = edges[e], d = edges[NEDGES + e];
  float x = coords[d * 3 + 0] - coords[s * 3 + 0];
  float y = coords[d * 3 + 1] - coords[s * 3 + 1];
  float z = coords[d * 3 + 2] - coords[s * 3 + 2];
  float len = sqrtf(x * x + y * y + z * z);
  float il = 1.f / (len + 1e-9f);
  float ux = x * il, uy = y * il, uz = z * il;
  const float s3 = 1.7320508f, s15 = 3.8729833f, s5 = 2.236068f;
  float* shp = sh + (size_t)e * SHD;
  shp[0] = 1.f;
  shp[1] = s3 * ux;  shp[2] = s3 * uy;  shp[3] = s3 * uz;
  shp[4] = s15 * ux * uy;  shp[5] = s15 * uy * uz;
  shp[6] = 0.5f * s5 * (3.f * uz * uz - 1.f);
  shp[7] = s15 * ux * uz;  shp[8] = 0.5f * s15 * (ux * ux - uy * uy);
  const float gap = MAXLEN / (EDGESZ - 1);
  _Float16* gep = ge + (size_t)e * EDGESZ;
#pragma unroll 4
  for (int o = 0; o < EDGESZ; ++o) {
    float t = (len - gap * (float)o) / gap;
    gep[o] = (_Float16)__expf(-0.5f * t * t);
  }
}

// ---------------------------------------------------------------------------
// h init: h[:, :48] = node_attr, rest 0
// ---------------------------------------------------------------------------
__global__ void init_h_kernel(const float* __restrict__ node_attr,
                              float* __restrict__ h) {
  long long stride = (long long)gridDim.x * blockDim.x;
  for (long long i = blockIdx.x * (long long)blockDim.x + threadIdx.x;
       i < (long long)NNODES * HSTRIDE; i += stride) {
    int n = (int)(i / HSTRIDE), c = (int)(i % HSTRIDE);
    h[i] = (c < NS) ? node_attr[(size_t)n * NS + c] : 0.f;
  }
}

// ---------------------------------------------------------------------------
// Edge counts per src node (segment-mean denominator)
// ---------------------------------------------------------------------------
__global__ void count_kernel(const int* __restrict__ src, int* __restrict__ cnt) {
  int e = blockIdx.x * blockDim.x + threadIdx.x;
  if (e < NEDGES) atomicAdd(&cnt[src[e]], 1);
}

// ---------------------------------------------------------------------------
// Gather edge features ea = [emb | edge_type | h[src,:48] | h[dst,:48]] (f16)
// ---------------------------------------------------------------------------
__global__ void gather_ea_kernel(const _Float16* __restrict__ emb16,
                                 const _Float16* __restrict__ et16,
                                 const float* __restrict__ h,
                                 const int* __restrict__ src,
                                 const int* __restrict__ dst,
                                 _Float16* __restrict__ ea) {
  long long stride = (long long)gridDim.x * blockDim.x;
  for (long long i = blockIdx.x * (long long)blockDim.x + threadIdx.x;
       i < (long long)NEDGES * NEF; i += stride) {
    int e = (int)(i / NEF), c = (int)(i % NEF);
    _Float16 v;
    if (c < 64)       v = emb16[(size_t)e * 64 + c];
    else if (c < 128) v = et16[(size_t)e * 64 + (c - 64)];
    else if (c < 176) v = (_Float16)h[(size_t)src[e] * HSTRIDE + (c - 128)];
    else              v = (_Float16)h[(size_t)dst[e] * HSTRIDE + (c - 176)];
    ea[i] = v;
  }
}

// ---------------------------------------------------------------------------
// Fused per-layer conv, TDM-staged weights. Per wave: 16 edges.
//   Weight column tiles (224x16 f16) are DMA'd into a double-buffered LDS
//   stage by wave 0 (tensor_load_to_lds / s_wait_tensorcnt), overlapping the
//   WMMA of the previous tile; both waves read B fragments from LDS.
//   GEMM1: hidden = relu(ea @ fc1 + bc1)                 -> LDS f16
//   GEMM2: fc2 tiles, epilogue contracts sh into msg     -> LDS f32 (ds_add)
//   msg *= h[dst]; GEMM3: msg @ Wout -> atomicAdd accum[src]
// ---------------------------------------------------------------------------
#define WPB 2       // waves per block
#define LDH 232     // padded LDS row stride (halves)

__global__ void conv_fused_kernel(const _Float16* __restrict__ ea,
                                  const float* __restrict__ sh,
                                  const float* __restrict__ h,
                                  const int* __restrict__ src,
                                  const int* __restrict__ dst,
                                  const _Float16* __restrict__ fc1,
                                  const float* __restrict__ bc1,
                                  const _Float16* __restrict__ fc2,
                                  const float* __restrict__ bc2,
                                  const _Float16* __restrict__ Wout,
                                  float* __restrict__ accum,
                                  int din, int dout) {
  __shared__ _Float16 s_hid[WPB][16][LDH];
  __shared__ float    s_msg[WPB][16][176];
  __shared__ float    s_shh[WPB][16][SHD];
  __shared__ _Float16 s_w2[2][NEF * 16];   // double-buffered TDM weight stage

  int w    = threadIdx.x >> 5;
  int lane = threadIdx.x & 31;
  int e0   = (blockIdx.x * WPB + w) << 4;
  int n_   = lane & 15, hi = lane >> 4;

  unsigned wbuf_off[2];
  wbuf_off[0] = (unsigned)(unsigned long long)&s_w2[0][0];
  wbuf_off[1] = (unsigned)(unsigned long long)&s_w2[1][0];

  // stage sh + zero msg accumulator
  for (int idx = lane; idx < 16 * SHD; idx += 32)
    s_shh[w][idx / SHD][idx % SHD] = sh[(size_t)(e0 + idx / SHD) * SHD + idx % SHD];
  for (int idx = lane; idx < 16 * 176; idx += 32)
    s_msg[w][idx / 176][idx % 176] = 0.f;

  // ---- GEMM1: hidden(16x224) = relu(ea_tile @ fc1 + bc1), fc1 via TDM ----
  if (w == 0) tdm_load_w_tile(fc1, NEF, 0, wbuf_off[0], NEF);
  for (int nt = 0; nt < NEF / 16; ++nt) {
    if (w == 0) {
      tdm_wait0();                                      // tile nt landed
      if (nt + 1 < NEF / 16)                            // prefetch tile nt+1
        tdm_load_w_tile(fc1, NEF, (nt + 1) * 16, wbuf_off[(nt + 1) & 1], NEF);
    }
    __syncthreads();
    v8f acc = {};
    for (int k0 = 0; k0 < NEF; k0 += 32) {
      v16h a = load_a_frag(ea, NEF, e0, k0, NEF);
      v16h b = load_b_frag(&s_w2[nt & 1][0], 16, k0, 0, NEF, 16);
      acc = wmma_f16(a, b, acc);
    }
    int col = nt * 16 + n_;
    float bb = bc1[col];
#pragma unroll
    for (int r = 0; r < 8; ++r) {
      float v = acc[r] + bb;
      s_hid[w][r + 8 * hi][col] = (_Float16)(v > 0.f ? v : 0.f);
    }
    __syncthreads();
  }

  // ---- GEMM2 + spherical-harmonic contraction, fc2 tiles via TDM ----
  int N2 = din * SHD;
  int tN2 = N2 / 16;
  if (w == 0) tdm_load_w_tile(fc2, N2, 0, wbuf_off[0], NEF);
  for (int nt = 0; nt < tN2; ++nt) {
    if (w == 0) {
      tdm_wait0();
      if (nt + 1 < tN2)
        tdm_load_w_tile(fc2, N2, (nt + 1) * 16, wbuf_off[(nt + 1) & 1], NEF);
    }
    __syncthreads();
    v8f acc = {};
    for (int k0 = 0; k0 < NEF; k0 += 32) {
      v16h a = load_a_frag(&s_hid[w][0][0], LDH, 0, k0, NEF);
      v16h b = load_b_frag(&s_w2[nt & 1][0], 16, k0, 0, NEF, 16);
      acc = wmma_f16(a, b, acc);
    }
    int col = nt * 16 + n_;
    float bb = bc2[col];
    int i = col / SHD, s = col % SHD;
#pragma unroll
    for (int r = 0; r < 8; ++r) {
      int m = r + 8 * hi;
      atomicAdd(&s_msg[w][m][i], (acc[r] + bb) * s_shh[w][m][s]);  // ds_add_f32
    }
    __syncthreads();
  }

  // ---- multiply by gathered h[dst], repack f16 for GEMM3 ----
  for (int idx = lane; idx < 16 * din; idx += 32) {
    int m = idx / din, i = idx % din;
    float hv = h[(size_t)dst[e0 + m] * HSTRIDE + i];
    s_hid[w][m][i] = (_Float16)(s_msg[w][m][i] * hv);
  }
  __syncthreads();

  // ---- GEMM3: msg(16xdin) @ Wout(din x dout) -> scatter to accum[src] ----
  int kend = (din + 31) & ~31;
  for (int nt = 0; nt < dout / 16; ++nt) {
    v8f acc = {};
    for (int k0 = 0; k0 < kend; k0 += 32) {
      v16h a = load_a_frag(&s_hid[w][0][0], LDH, 0, k0, din);
      v16h b = load_b_frag(Wout, dout, k0, nt * 16, din, dout);
      acc = wmma_f16(a, b, acc);
    }
    int col = nt * 16 + n_;
#pragma unroll
    for (int r = 0; r < 8; ++r) {
      int e = e0 + r + 8 * hi;
      atomicAdd(&accum[(size_t)src[e] * dout + col], acc[r]);
    }
  }
}

// ---------------------------------------------------------------------------
// Per-node: segment mean, layernorm (gamma/beta), padded residual into h
// ---------------------------------------------------------------------------
__global__ void node_update_kernel(const float* __restrict__ accum,
                                   const int* __restrict__ cnt,
                                   const float* __restrict__ gamma,
                                   const float* __restrict__ beta,
                                   float* __restrict__ h, int dout) {
  int n = blockIdx.x * blockDim.x + threadIdx.x;
  if (n >= NNODES) return;
  float c = (float)cnt[n]; if (c < 1.f) c = 1.f;
  float inv = 1.f / c;
  float sum = 0.f, sum2 = 0.f;
  const float* ap = accum + (size_t)n * dout;
  for (int j = 0; j < dout; ++j) {
    float v = ap[j] * inv;
    sum += v; sum2 += v * v;
  }
  float mu = sum / (float)dout;
  float var = sum2 / (float)dout - mu * mu;
  float rs = rsqrtf(var + 1e-5f);
  float* hp = h + (size_t)n * HSTRIDE;
  for (int j = 0; j < dout; ++j) {
    float v = ap[j] * inv;
    hp[j] += (v - mu) * rs * gamma[j] + beta[j];
  }
}

// ---------------------------------------------------------------------------
// Gather ne = [h[:, :48] | h[:, -48:]] as f16 (final h has 224 cols)
// ---------------------------------------------------------------------------
__global__ void gather_ne_kernel(const float* __restrict__ h,
                                 _Float16* __restrict__ ne) {
  long long stride = (long long)gridDim.x * blockDim.x;
  for (long long i = blockIdx.x * (long long)blockDim.x + threadIdx.x;
       i < (long long)NNODES * 96; i += stride) {
    int n = (int)(i / 96), c = (int)(i % 96);
    float v = (c < NS) ? h[(size_t)n * HSTRIDE + c]
                       : h[(size_t)n * HSTRIDE + 176 + (c - NS)];
    ne[i] = (_Float16)v;
  }
}

// ---------------------------------------------------------------------------
// Host: workspace layout + launch sequence
// ---------------------------------------------------------------------------
static const int kDin[3]  = {48, 112, 176};
static const int kDout[3] = {112, 176, 224};

extern "C" void kernel_launch(void* const* d_in, const int* in_sizes, int n_in,
                              void* d_out, int out_size, void* d_ws, size_t ws_size,
                              hipStream_t stream) {
  // ---- input pointers (setup_inputs dict order, params flattened in order) ----
  const float* node_attr = (const float*)d_in[0];
  const float* coords    = (const float*)d_in[1];
  // d_in[2] batch_id, d_in[3] perturb_mask: unused by the reference
  const int*   edges     = (const int*)d_in[4];
  const float* etype     = (const float*)d_in[5];
  const float* We1 = (const float*)d_in[6];  const float* be1 = (const float*)d_in[7];
  const float* We2 = (const float*)d_in[8];  const float* be2 = (const float*)d_in[9];
  const float* Wf1 = (const float*)d_in[10]; const float* bf1 = (const float*)d_in[11];
  const float* Wf2 = (const float*)d_in[12]; const float* bf2 = (const float*)d_in[13];
  const float *fc1[3], *bc1[3], *fc2[3], *bc2[3], *WoutF[3], *gamma[3], *beta[3];
  for (int l = 0; l < 3; ++l) {
    int b = 14 + 7 * l;
    fc1[l]   = (const float*)d_in[b + 0];
    bc1[l]   = (const float*)d_in[b + 1];
    fc2[l]   = (const float*)d_in[b + 2];
    bc2[l]   = (const float*)d_in[b + 3];
    WoutF[l] = (const float*)d_in[b + 4];
    gamma[l] = (const float*)d_in[b + 5];
    beta[l]  = (const float*)d_in[b + 6];
  }
  const int* src = edges;
  const int* dst = edges + NEDGES;

  // ---- carve workspace ----
  char* ws = (char*)d_ws;
  size_t off = 0;
  auto carve = [&](size_t bytes) -> char* {
    char* p = ws + off;
    off = (off + bytes + 255) & ~(size_t)255;
    return p;
  };
  float*    sh     = (float*)   carve((size_t)NEDGES * SHD * 4);
  _Float16* ge16   = (_Float16*)carve((size_t)NEDGES * 64 * 2);
  _Float16* t64    = (_Float16*)carve((size_t)NEDGES * 64 * 2);
  _Float16* emb16  = (_Float16*)carve((size_t)NEDGES * 64 * 2);
  _Float16* et16   = (_Float16*)carve((size_t)NEDGES * 64 * 2);
  _Float16* ea16   = (_Float16*)carve((size_t)NEDGES * NEF * 2);
  float*    hbuf   = (float*)   carve((size_t)NNODES * HSTRIDE * 4);
  float*    accum  = (float*)   carve((size_t)NNODES * HSTRIDE * 4);
  int*      cnt    = (int*)     carve((size_t)NNODES * 4);
  _Float16* ne16   = (_Float16*)carve((size_t)NNODES * 96 * 2);
  _Float16* nh16   = (_Float16*)carve((size_t)NNODES * 96 * 2);
  _Float16* We1h   = (_Float16*)carve(64 * 64 * 2);
  _Float16* We2h   = (_Float16*)carve(64 * 64 * 2);
  _Float16* Wf1h   = (_Float16*)carve(96 * 96 * 2);
  _Float16* Wf2h   = (_Float16*)carve(96 * 48 * 2);
  _Float16 *fc1h[3], *fc2h[3], *wouth[3];
  for (int l = 0; l < 3; ++l) {
    fc1h[l]  = (_Float16*)carve((size_t)NEF * NEF * 2);
    fc2h[l]  = (_Float16*)carve((size_t)NEF * kDin[l] * SHD * 2);
    wouth[l] = (_Float16*)carve((size_t)kDin[l] * kDout[l] * 2);
  }
  (void)ws_size; (void)in_sizes; (void)n_in; (void)out_size;

  auto cvt = [&](const float* in, _Float16* outp, long long n) {
    int blocks = (int)((n + 255) / 256); if (blocks > 2048) blocks = 2048;
    hipLaunchKernelGGL(cvt16_kernel, dim3(blocks), dim3(256), 0, stream, in, outp, n);
  };

  // ---- weight / feature conversion ----
  cvt(We1, We1h, 64 * 64);  cvt(We2, We2h, 64 * 64);
  cvt(Wf1, Wf1h, 96 * 96);  cvt(Wf2, Wf2h, 96 * 48);
  cvt(etype, et16, (long long)NEDGES * 64);
  for (int l = 0; l < 3; ++l) {
    cvt(fc1[l], fc1h[l], (long long)NEF * NEF);
    cvt(fc2[l], fc2h[l], (long long)NEF * kDin[l] * SHD);
    cvt(WoutF[l], wouth[l], (long long)kDin[l] * kDout[l]);
  }

  // ---- edge geometry ----
  hipLaunchKernelGGL(edge_geom_kernel, dim3(NEDGES / 256), dim3(256), 0, stream,
                     coords, edges, sh, ge16);

  // ---- radial embedding MLP (two WMMA GEMMs) ----
  {
    int waves = (NEDGES / 16) * (64 / 16);
    int blocks = (waves + 7) / 8;
    hipLaunchKernelGGL((wgemm_kernel<true, true>), dim3(blocks), dim3(256), 0, stream,
                       ge16, We1h, be1, t64, (float*)nullptr, NEDGES, 64, 64);
    hipLaunchKernelGGL((wgemm_kernel<false, true>), dim3(blocks), dim3(256), 0, stream,
                       t64, We2h, be2, emb16, (float*)nullptr, NEDGES, 64, 64);
  }

  // ---- node state + segment counts ----
  hipLaunchKernelGGL(init_h_kernel, dim3(2048), dim3(256), 0, stream, node_attr, hbuf);
  hipMemsetAsync(cnt, 0, (size_t)NNODES * 4, stream);
  hipLaunchKernelGGL(count_kernel, dim3(NEDGES / 256), dim3(256), 0, stream, src, cnt);

  // ---- three interaction layers ----
  for (int l = 0; l < 3; ++l) {
    hipLaunchKernelGGL(gather_ea_kernel, dim3(2048), dim3(256), 0, stream,
                       emb16, et16, hbuf, src, dst, ea16);
    hipMemsetAsync(accum, 0, (size_t)NNODES * kDout[l] * 4, stream);
    hipLaunchKernelGGL(conv_fused_kernel, dim3(NEDGES / 16 / WPB), dim3(32 * WPB), 0,
                       stream, ea16, sh, hbuf, src, dst,
                       fc1h[l], bc1[l], fc2h[l], bc2[l], wouth[l],
                       accum, kDin[l], kDout[l]);
    hipLaunchKernelGGL(node_update_kernel, dim3(NNODES / 256), dim3(256), 0, stream,
                       accum, cnt, gamma[l], beta[l], hbuf, kDout[l]);
  }

  // ---- final node MLP: [h[:,:48] | h[:,-48:]] -> 96 -> relu -> 48 ----
  hipLaunchKernelGGL(gather_ne_kernel, dim3(768), dim3(256), 0, stream, hbuf, ne16);
  {
    int waves1 = (NNODES / 16) * (96 / 16);
    hipLaunchKernelGGL((wgemm_kernel<true, true>), dim3((waves1 + 7) / 8), dim3(256), 0,
                       stream, ne16, Wf1h, bf1, nh16, (float*)nullptr, NNODES, 96, 96);
    int waves2 = (NNODES / 16) * (48 / 16);
    hipLaunchKernelGGL((wgemm_kernel<false, false>), dim3((waves2 + 7) / 8), dim3(256), 0,
                       stream, nh16, Wf2h, bf2, (_Float16*)nullptr, (float*)d_out,
                       NNODES, 48, 96);
  }
}
